// BlockwiseResonatorModel_66520453480531
// MI455X (gfx1250) — matrix operand order repaired and checked by reading, not compiled, not run on hardware
//
#include <hip/hip_runtime.h>
#include <hip/hip_bf16.h>
#include <math.h>

// ---------------------------------------------------------------------------
// BlockwiseResonatorModel on gfx1250 (MI455X).
//  - conv towers as implicit GEMM with v_wmma_f32_16x16x32_bf16
//  - 32KB LDS B-panels (barrier-free inner K loop), async loads to LDS when
//    the gfx1250 builtin is available
//  - Gumbel straight-through one-hot  => final FFT-conv collapses to a shift
//  - 512-pt radix-2 FFT/iFFT in LDS, resonator scan, overlap-add
// ---------------------------------------------------------------------------

typedef __attribute__((ext_vector_type(16))) __bf16 v16bf;
typedef __attribute__((ext_vector_type(8)))  float  v8f;

#define NSAMP 131072
#define NB    128
#define CHN   512
#define NCB   257

#if __has_builtin(__builtin_amdgcn_global_load_async_to_lds_b128)
#define HAVE_ASYNC_LDS 1
typedef int i32x4 __attribute__((__vector_size__(16)));
typedef __attribute__((address_space(1))) i32x4 GI4;   // global int4
typedef __attribute__((address_space(3))) i32x4 LI4;   // LDS int4
static __device__ __forceinline__ void async_ld128(const void* g, void* l) {
  __builtin_amdgcn_global_load_async_to_lds_b128((GI4*)g, (LI4*)l, 0, 0);
}
static __device__ __forceinline__ void wait_async0() {
#if __has_builtin(__builtin_amdgcn_s_wait_asynccnt)
  __builtin_amdgcn_s_wait_asynccnt(0);
#else
  asm volatile("s_wait_asynccnt 0x0" ::: "memory");
#endif
}
#endif

// ------------------------------- converters --------------------------------

__global__ __launch_bounds__(256) void cvt_bf16_k(const float* __restrict__ in,
                                                  __bf16* __restrict__ out, int n) {
  int i = blockIdx.x * 256 + threadIdx.x;
  if (i < n) out[i] = (__bf16)in[i];
}

// w0 (512,4096) f32 -> w0t (4096,512) bf16
__global__ __launch_bounds__(256) void cvt_w0t_k(const float* __restrict__ w0,
                                                 __bf16* __restrict__ o) {
  int t = blockIdx.x * 256 + threadIdx.x;     // t < 4096*512
  int n = t >> 9, k = t & 511;
  o[t] = (__bf16)w0[(size_t)k * 4096 + n];
}

// w (512co,512ci,KK) f32 -> (KK,512co,512ci) bf16
__global__ __launch_bounds__(256) void cvt_wl_k(const float* __restrict__ w,
                                                __bf16* __restrict__ o, int KK) {
  int t = blockIdx.x * 256 + threadIdx.x;
  if (t >= (KK << 18)) return;
  int ci = t & 511, co = (t >> 9) & 511, k = t >> 18;
  o[t] = (__bf16)w[((size_t)(co * 512 + ci)) * KK + k];
}

// wf_tr (257,512,3) f32 -> (3,320,512) bf16, zero-padded co
__global__ __launch_bounds__(256) void cvt_wftr_k(const float* __restrict__ wf,
                                                  __bf16* __restrict__ o) {
  int t = blockIdx.x * 256 + threadIdx.x;
  if (t >= 3 * 320 * 512) return;
  int ci = t & 511; int rem = t >> 9; int co = rem % 320; int k = rem / 320;
  o[t] = (co < 257) ? (__bf16)wf[((size_t)(co * 512 + ci)) * 3 + k] : (__bf16)0.0f;
}

// ------------------------------ activation prep ----------------------------
// mode 0: linear 2x upsample (half-pixel, edge clamped) + zero pad 1  (Lp=2L+2)
// mode 1: transpose-conv dilation (insert zeros) + zero pad 2         (Lp=2L+3)
// mode 2: zero pad 1                                                  (Lp=L+2)
__global__ __launch_bounds__(256) void prepare_u_k(const __bf16* __restrict__ H,
                                                   __bf16* __restrict__ U,
                                                   int L, int Lp, int mode) {
  size_t idx = (size_t)blockIdx.x * 256 + threadIdx.x;
  size_t total = (size_t)NB * Lp * CHN;
  if (idx >= total) return;
  int c = (int)(idx & 511);
  int p = (int)((idx >> 9) % (size_t)Lp);
  int b = (int)((idx >> 9) / (size_t)Lp);
  float v = 0.f;
  if (mode == 0) {
    int q = p - 1;
    if (q >= 0 && q < 2 * L) {
      int j = q >> 1;
      int jn = j + ((q & 1) ? 1 : -1);
      jn = jn < 0 ? 0 : (jn >= L ? L - 1 : jn);
      v = 0.75f * (float)H[((size_t)b * L + j) * CHN + c] +
          0.25f * (float)H[((size_t)b * L + jn) * CHN + c];
    }
  } else if (mode == 1) {
    int q = p - 2;
    if (q >= 0 && !(q & 1)) {
      int j = q >> 1;
      if (j < L) v = (float)H[((size_t)b * L + j) * CHN + c];
    }
  } else {
    int q = p - 1;
    if (q >= 0 && q < L) v = (float)H[((size_t)b * L + q) * CHN + c];
  }
  U[idx] = (__bf16)v;
}

// ----------------------- shared LDS panel fill helper ----------------------
// Fill 64(co) x 128(ci) bf16 (16KB) into the 64x256 panel from Wsrc (row stride CHN bf16).
static __device__ __forceinline__ void fill_panel(__bf16* ldsB,
                                                  const __bf16* __restrict__ Wsrc,
                                                  int tid) {
#pragma unroll
  for (int u = 0; u < 4; ++u) {
    const int q = u * 256 + tid;          // 0..1023 chunks of 8 bf16
    const int co = q >> 4, ch = q & 15;
    const __bf16* g = Wsrc + (size_t)co * CHN + ch * 8;
    __bf16* l = &ldsB[co * 256 + ch * 8];
#if defined(HAVE_ASYNC_LDS)
    async_ld128(g, l);
#else
    *(uint4*)l = *(const uint4*)g;
#endif
  }
#if defined(HAVE_ASYNC_LDS)
  wait_async0();
#endif
}

// ------------------------------- WMMA GEMMs --------------------------------
// h0 = x @ w0 + b0, scattered to NLC (B,8,512).  M=128 (batch), N=4096, K=512.
__global__ __launch_bounds__(256) void gemm_h0_k(const __bf16* __restrict__ X,
                                                 const __bf16* __restrict__ W0T,
                                                 const float* __restrict__ b0,
                                                 __bf16* __restrict__ H) {
  __shared__ __attribute__((aligned(16))) __bf16 ldsB[64 * 256];
  const int tid = threadIdx.x;
  const int wave = tid >> 5, lane = tid & 31;
  const int coTile = blockIdx.x << 6;
  const int bRow = wave * 16 + (lane & 15);
  const int kbA = (lane >> 4) << 3;
  const int kbB = (lane >> 4) << 3;
  v8f acc[4] = {};
  for (int half = 0; half < 2; ++half) {
    __syncthreads();
    fill_panel(ldsB, W0T + (size_t)coTile * CHN + half * 256, tid);
    __syncthreads();
    const __bf16* Arow = X + (size_t)bRow * CHN + half * 256;
#pragma unroll 2
    for (int kc = 0; kc < 8; ++kc) {
      union F { v16bf v; uint4 q[2]; };
      F af, bf4[4];
      af.q[0] = *(const uint4*)(Arow + kc * 32 + kbA);
      af.q[1] = *(const uint4*)(Arow + kc * 32 + kbA + 16);
#pragma unroll
      for (int j = 0; j < 4; ++j) {
        const int cr = j * 16 + (lane & 15);
        bf4[j].q[0] = *(const uint4*)&ldsB[cr * 256 + kc * 32 + kbB];
        bf4[j].q[1] = *(const uint4*)&ldsB[cr * 256 + kc * 32 + kbB + 16];
      }
#pragma unroll
      for (int j = 0; j < 4; ++j)
        acc[j] = __builtin_amdgcn_wmma_f32_16x16x32_bf16(
            false, af.v, false, bf4[j].v, (short)0, acc[j], false, false);
    }
  }
  const int rHi = (lane >> 4) << 3;
#pragma unroll
  for (int j = 0; j < 4; ++j) {
    const int col = coTile + j * 16 + (lane & 15);
    const int c = col >> 3, s = col & 7;
    const float bv = b0[col];
#pragma unroll
    for (int r = 0; r < 8; ++r) {
      const int b = wave * 16 + rHi + r;
      H[((size_t)b * 8 + s) * CHN + c] = (__bf16)(acc[j][r] + bv);
    }
  }
}

// Implicit-GEMM conv over u (B,Lp,512) with weights (KK,COpad,512).
// mode 0: leaky(0.2) -> Hout bf16 (B,L,512)
// mode 1: sigmoid^2  -> Tout f32  (B,L,257)   (COpad=320, COreal=257)
__global__ __launch_bounds__(256) void conv_gemm_k(const __bf16* __restrict__ U,
                                                   const __bf16* __restrict__ Wb,
                                                   const float* __restrict__ bias,
                                                   __bf16* __restrict__ Hout,
                                                   float* __restrict__ Tout,
                                                   int Lp, int lg, int KK,
                                                   int COpad, int COreal, int mode) {
  __shared__ __attribute__((aligned(16))) __bf16 ldsB[64 * 256];
  const int tid = threadIdx.x;
  const int wave = tid >> 5, lane = tid & 31;
  const int coTile = blockIdx.x << 6;
  const int mBlock = blockIdx.y << 7;
  const int L = 1 << lg;
  const int mRow = mBlock + wave * 16 + (lane & 15);
  const int bb = mRow >> lg;
  const int ll = mRow & (L - 1);
  const int kbA = (lane >> 4) << 3;
  const int kbB = (lane >> 4) << 3;
  v8f acc[4] = {};
  for (int k = 0; k < KK; ++k) {
    const __bf16* Arow = U + ((size_t)bb * Lp + ll + k) * CHN;
    for (int half = 0; half < 2; ++half) {
      __syncthreads();
      fill_panel(ldsB, Wb + (size_t)(k * COpad + coTile) * CHN + half * 256, tid);
      __syncthreads();
#pragma unroll 2
      for (int kc = 0; kc < 8; ++kc) {
        union F { v16bf v; uint4 q[2]; };
        F af, bf4[4];
        const int aoff = half * 256 + kc * 32;
        af.q[0] = *(const uint4*)(Arow + aoff + kbA);
        af.q[1] = *(const uint4*)(Arow + aoff + kbA + 16);
#pragma unroll
        for (int j = 0; j < 4; ++j) {
          const int cr = j * 16 + (lane & 15);
          bf4[j].q[0] = *(const uint4*)&ldsB[cr * 256 + kc * 32 + kbB];
          bf4[j].q[1] = *(const uint4*)&ldsB[cr * 256 + kc * 32 + kbB + 16];
        }
#pragma unroll
        for (int j = 0; j < 4; ++j)
          acc[j] = __builtin_amdgcn_wmma_f32_16x16x32_bf16(
              false, af.v, false, bf4[j].v, (short)0, acc[j], false, false);
      }
    }
  }
  const int rHi = (lane >> 4) << 3;
#pragma unroll
  for (int j = 0; j < 4; ++j) {
    const int co = coTile + j * 16 + (lane & 15);
    const float bv = bias[co < COreal ? co : 0];
#pragma unroll
    for (int r = 0; r < 8; ++r) {
      const int m = mBlock + wave * 16 + rHi + r;
      const int b2 = m >> lg, l2 = m & (L - 1);
      float v = acc[j][r] + bv;
      if (mode == 0) {
        v = v >= 0.f ? v : 0.2f * v;
        Hout[((size_t)b2 * L + l2) * CHN + co] = (__bf16)v;
      } else if (co < COreal) {
        const float sg = 1.f / (1.f + __expf(-v));
        Tout[((size_t)b2 * L + l2) * NCB + co] = sg * sg;
      }
    }
  }
}

// ------------------------- single-channel final conv -----------------------
// out[b,l] = bf + sum_{k,ci} wf[ci,k] * u[b,l+k,ci];  mode 0: ()^2  mode 1: +gumbel
__global__ __launch_bounds__(256) void final_dot1_k(const __bf16* __restrict__ U,
                                                    const float* __restrict__ wf,
                                                    const float* __restrict__ bf,
                                                    const float* __restrict__ gum,
                                                    float* __restrict__ outv, int mode) {
  const int gid = blockIdx.x * 8 + (threadIdx.x >> 5);   // one wave per (b,l)
  const int lane = threadIdx.x & 31;
  const int b = gid >> 9, l = gid & 511;
  const __bf16* base = U + ((size_t)b * 514 + l) * CHN;
  float s = 0.f;
#pragma unroll
  for (int k = 0; k < 3; ++k)
    for (int ci = lane; ci < CHN; ci += 32)
      s += (float)base[k * CHN + ci] * wf[ci * 3 + k];
  for (int off = 16; off; off >>= 1) s += __shfl_down(s, off, 32);
  if (lane == 0) {
    float v = s + bf[0];
    if (mode == 0) outv[gid] = v * v;
    else           outv[gid] = v + (-__logf(-__logf(gum[gid])));
  }
}

__global__ __launch_bounds__(256) void argmax_k(const float* __restrict__ ll,
                                                int* __restrict__ sb) {
  const int b = blockIdx.x, t = threadIdx.x;
  __shared__ float sv[256]; __shared__ int si[256];
  float v0 = ll[b * 512 + t], v1 = ll[b * 512 + t + 256];
  float v; int ii;
  if (v0 >= v1) { v = v0; ii = t; } else { v = v1; ii = t + 256; }
  sv[t] = v; si[t] = ii; __syncthreads();
  for (int s = 128; s; s >>= 1) {
    if (t < s && sv[t + s] > sv[t]) { sv[t] = sv[t + s]; si[t] = si[t + s]; }
    __syncthreads();
  }
  if (t == 0) sb[b] = si[0] * 256;
}

// ------------------------------ spectral stage -----------------------------
__global__ __launch_bounds__(256) void fft_fwd_k(const float* __restrict__ impsq,
                                                 const float* __restrict__ noise,
                                                 float2* __restrict__ freq) {
  const int fr = blockIdx.x & 511, b = blockIdx.x >> 9;
  __shared__ float re[512], im[512];
  const int t = threadIdx.x;
  for (int jj = 0; jj < 2; ++jj) {
    const int j = t + jj * 256;
    const int n = fr * 256 + j;
    float v = 0.f;
    if (n < NSAMP) {
      float coord = (float)n * (1.f / 256.f) + (1.f / 512.f) - 0.5f;
      int i0 = (int)floorf(coord);
      float fr2 = coord - (float)i0;
      int idx0 = i0 < 0 ? 0 : (i0 > 511 ? 511 : i0);
      int idx1 = (i0 + 1) < 0 ? 0 : ((i0 + 1) > 511 ? 511 : (i0 + 1));
      float a = impsq[b * 512 + idx0], c = impsq[b * 512 + idx1];
      v = (a + (c - a) * fr2) * noise[n];
    }
    const float w = 0.54f - 0.46f * __cosf(6.28318530717958647f * (float)j / 512.f);
    const int rj = __brev((unsigned)j) >> 23;
    re[rj] = v * w; im[rj] = 0.f;
  }
  __syncthreads();
  for (int s = 0; s < 9; ++s) {
    const int half = 1 << s;
    const int grp = t >> s, pos = t & (half - 1);
    const int i = grp * (half << 1) + pos, j2 = i + half;
    float wr, wi;
    __sincosf(-6.28318530717958647f * (float)pos / (float)(half << 1), &wi, &wr);
    const float xr = re[j2], xi = im[j2];
    const float tr = xr * wr - xi * wi, ti = xr * wi + xi * wr;
    const float ur = re[i], ui = im[i];
    re[i] = ur + tr; im[i] = ui + ti;
    re[j2] = ur - tr; im[j2] = ui - ti;
    __syncthreads();
  }
  const float sc = 0.0441941738241592f;  // 1/sqrt(512)
  for (int c = t; c < NCB; c += 256) {
    float2 o; o.x = re[c] * sc; o.y = im[c] * sc;
    freq[((size_t)b * 512 + fr) * NCB + c] = o;
  }
}

// resonator scan: spec[fr] = freq[fr] + t[fr] * spec[fr-1]  (in place)
__global__ __launch_bounds__(256) void scan_res_k(float2* __restrict__ freq,
                                                  const float* __restrict__ tco) {
  const int idx = blockIdx.x * 256 + threadIdx.x;
  if (idx >= NB * NCB) return;
  const int b = idx / NCB, c = idx % NCB;
  const size_t base = (size_t)b * 512 * NCB + c;
  float cx = 0.f, cy = 0.f;
#pragma unroll 4
  for (int fr = 0; fr < 512; ++fr) {
    const float2 f = freq[base + (size_t)fr * NCB];
    const float tv = tco[base + (size_t)fr * NCB];
    cx = f.x + tv * cx; cy = f.y + tv * cy;
    float2 o; o.x = cx; o.y = cy;
    freq[base + (size_t)fr * NCB] = o;
  }
}

// inverse: Hermitian-extend, inverse FFT (ortho), take real part.
__global__ __launch_bounds__(256) void fft_inv_k(const float2* __restrict__ spec,
                                                 float* __restrict__ frames) {
  const int fr = blockIdx.x & 511, b = blockIdx.x >> 9;
  __shared__ float re[512], im[512];
  const int t = threadIdx.x;
  const size_t base = ((size_t)b * 512 + fr) * NCB;
  for (int jj = 0; jj < 2; ++jj) {
    const int c = t + jj * 256;
    float xr, xi;
    if (c <= 256) { float2 v = spec[base + c]; xr = v.x; xi = v.y; }
    else          { float2 v = spec[base + (512 - c)]; xr = v.x; xi = -v.y; }
    const int rc = __brev((unsigned)c) >> 23;
    re[rc] = xr; im[rc] = xi;
  }
  __syncthreads();
  for (int s = 0; s < 9; ++s) {
    const int half = 1 << s;
    const int grp = t >> s, pos = t & (half - 1);
    const int i = grp * (half << 1) + pos, j2 = i + half;
    float wr, wi;
    __sincosf(6.28318530717958647f * (float)pos / (float)(half << 1), &wi, &wr);
    const float xr = re[j2], xi = im[j2];
    const float tr = xr * wr - xi * wi, ti = xr * wi + xi * wr;
    const float ur = re[i], ui = im[i];
    re[i] = ur + tr; im[i] = ui + ti;
    re[j2] = ur - tr; im[j2] = ui - ti;
    __syncthreads();
  }
  const float sc = 0.0441941738241592f;
  for (int jj = 0; jj < 2; ++jj) {
    const int n = t + jj * 256;
    frames[((size_t)b * 512 + fr) * 512 + n] = re[n] * sc;
  }
}

__global__ __launch_bounds__(256) void overlap_add_k(const float* __restrict__ frames,
                                                     float* __restrict__ sig) {
  const size_t idx = (size_t)blockIdx.x * 256 + threadIdx.x;
  if (idx >= (size_t)NB * NSAMP) return;
  const int b = (int)(idx >> 17);
  const int n = (int)(idx & (NSAMP - 1));
  const int s = n >> 8, i2 = n & 255;
  float v = frames[((size_t)b * 512 + s) * 512 + i2];
  if (s > 0) v += frames[((size_t)b * 512 + (s - 1)) * 512 + 256 + i2];
  sig[idx] = v;
}

// FFT-conv with one-hot location == shift by sb[b].
__global__ __launch_bounds__(256) void shift_out_k(const float* __restrict__ sig,
                                                   const int* __restrict__ sb,
                                                   float* __restrict__ out) {
  const size_t idx = (size_t)blockIdx.x * 256 + threadIdx.x;
  if (idx >= (size_t)NB * NSAMP) return;
  const int b = (int)(idx >> 17);
  const int n = (int)(idx & (NSAMP - 1));
  const int s0 = sb[b];
  out[idx] = (n >= s0) ? sig[((size_t)b << 17) + (n - s0)] : 0.f;
}

// ------------------------------- orchestration -----------------------------

extern "C" void kernel_launch(void* const* d_in, const int* in_sizes, int n_in,
                              void* d_out, int out_size, void* d_ws, size_t ws_size,
                              hipStream_t stream) {
  struct Net { const float *w0, *b0, *ws[6], *bs[6], *wf, *bf; int kk, wfsz; };
  Net nets[3]; int nn = 0;
  const float *xin = nullptr, *gum = nullptr, *noise = nullptr;
  {
    const float *vec1 = nullptr, *vec2 = nullptr;
    int i = 0;
    while (i < n_in) {
      const int sz = in_sizes[i];
      if (sz == 65536) { if (!vec1) vec1 = (const float*)d_in[i]; else vec2 = (const float*)d_in[i]; ++i; }
      else if (sz == 131072) { noise = (const float*)d_in[i]; ++i; }
      else if (sz == 2097152 && nn < 3) {          // insertion order: w0 b0 ws*6 bs*6 wf bf
        Net& N = nets[nn++];
        N.w0 = (const float*)d_in[i++]; N.b0 = (const float*)d_in[i++];
        const int wssz = in_sizes[i];
        for (int j = 0; j < 6; ++j) N.ws[j] = (const float*)d_in[i++];
        for (int j = 0; j < 6; ++j) N.bs[j] = (const float*)d_in[i++];
        N.wfsz = in_sizes[i]; N.wf = (const float*)d_in[i++];
        N.bf = (const float*)d_in[i++];
        N.kk = (wssz == 1048576) ? 4 : 3;
      }
      else if (sz == 4096 && nn < 3) {             // sorted pytree keys: b0 bf bs*6 w0 wf ws*6
        Net& N = nets[nn++];
        N.b0 = (const float*)d_in[i++];
        N.bf = (const float*)d_in[i++];
        for (int j = 0; j < 6; ++j) N.bs[j] = (const float*)d_in[i++];
        N.w0 = (const float*)d_in[i++];
        N.wfsz = in_sizes[i]; N.wf = (const float*)d_in[i++];
        const int wssz = in_sizes[i];
        for (int j = 0; j < 6; ++j) N.ws[j] = (const float*)d_in[i++];
        N.kk = (wssz == 1048576) ? 4 : 3;
      }
      else ++i;
    }
    xin = vec1; gum = vec2;   // setup_inputs dict order: x first, gumbel last
  }
  const Net *impN = nullptr, *trN = nullptr, *locN = nullptr;
  for (int j = 0; j < nn; ++j) {
    if (nets[j].kk == 4) locN = &nets[j];
    else if (nets[j].wfsz == 394752) trN = &nets[j];
    else impN = &nets[j];
  }
  if (!impN || !trN || !locN || !xin || !gum || !noise) return;
  const Net* order[3] = { impN, trN, locN };

  // ---- workspace bump allocator ----
  char* wsp = (char*)d_ws; size_t off = 0;
  auto alloc = [&](size_t bytes) -> void* {
    void* p = wsp + off; off += (bytes + 255) & ~(size_t)255; return p;
  };
  __bf16* xb = (__bf16*)alloc((size_t)65536 * 2);
  __bf16* w0t[3]; __bf16* wl[3][6];
  for (int nidx = 0; nidx < 3; ++nidx) {
    w0t[nidx] = (__bf16*)alloc((size_t)4096 * 512 * 2);
    for (int j = 0; j < 6; ++j)
      wl[nidx][j] = (__bf16*)alloc((size_t)order[nidx]->kk * 512 * 512 * 2);
  }
  __bf16* wfT   = (__bf16*)alloc((size_t)3 * 320 * 512 * 2);
  __bf16* Ubuf  = (__bf16*)alloc((size_t)NB * 514 * 512 * 2);   // frames aliases U+H
  __bf16* Hbuf  = (__bf16*)alloc((size_t)NB * 512 * 512 * 2);
  float*  tco   = (float*) alloc((size_t)NB * 512 * NCB * 4);
  float2* freq  = (float2*)alloc((size_t)NB * 512 * NCB * 8);
  float*  impsq = (float*) alloc((size_t)NB * 512 * 4);
  float*  loclg = (float*) alloc((size_t)NB * 512 * 4);
  int*    sbuf  = (int*)   alloc((size_t)NB * 4);
  float*  sig   = (float*) alloc((size_t)NB * NSAMP * 4);
  float*  frames = (float*)Ubuf;   // 134.2MB fits in U(67.4)+H(67.1)
  if (off > ws_size) return;

  const dim3 blk(256);
  // ---- weight / input conversion to bf16 ----
  cvt_bf16_k<<<dim3(65536 / 256), blk, 0, stream>>>(xin, xb, 65536);
  for (int nidx = 0; nidx < 3; ++nidx) {
    cvt_w0t_k<<<dim3(2097152 / 256), blk, 0, stream>>>(order[nidx]->w0, w0t[nidx]);
    const int kk = order[nidx]->kk;
    for (int j = 0; j < 6; ++j)
      cvt_wl_k<<<dim3(kk * 1024), blk, 0, stream>>>(order[nidx]->ws[j], wl[nidx][j], kk);
  }
  cvt_wftr_k<<<dim3(491520 / 256), blk, 0, stream>>>(trN->wf, wfT);

  // ---- run one conv tower; leaves padded final activation in Ubuf ----
  auto run_net = [&](int nidx, int umode) {
    gemm_h0_k<<<dim3(64, 1), blk, 0, stream>>>(xb, w0t[nidx], order[nidx]->b0, Hbuf);
    int L = 8;
    const int KK = order[nidx]->kk;
    for (int j = 0; j < 6; ++j) {
      const int L2 = L * 2;
      const int Lp = (umode == 0) ? (2 * L + 2) : (2 * L + 3);
      const size_t tot = (size_t)NB * Lp * 512;
      prepare_u_k<<<dim3((unsigned)((tot + 255) / 256)), blk, 0, stream>>>(Hbuf, Ubuf, L, Lp, umode);
      conv_gemm_k<<<dim3(8, L2), blk, 0, stream>>>(Ubuf, wl[nidx][j], order[nidx]->bs[j],
                                                   Hbuf, nullptr, Lp, 4 + j, KK, 512, 512, 0);
      L = L2;
    }
    const size_t tot = (size_t)NB * 514 * 512;
    prepare_u_k<<<dim3((unsigned)((tot + 255) / 256)), blk, 0, stream>>>(Hbuf, Ubuf, 512, 514, 2);
  };

  // imp tower -> imp^2 envelope (B,512)
  run_net(0, 0);
  final_dot1_k<<<dim3(8192), blk, 0, stream>>>(Ubuf, impN->wf, impN->bf, nullptr, impsq, 0);
  // tr tower -> sigmoid^2 transfer coefs (B,512,257)
  run_net(1, 0);
  conv_gemm_k<<<dim3(5, 512), blk, 0, stream>>>(Ubuf, wfT, trN->bf, nullptr, tco,
                                                514, 9, 3, 320, 257, 1);
  // loc tower (transpose-conv) -> gumbel logits -> hard argmax shift
  run_net(2, 1);
  final_dot1_k<<<dim3(8192), blk, 0, stream>>>(Ubuf, locN->wf, locN->bf, gum, loclg, 1);
  argmax_k<<<dim3(128), blk, 0, stream>>>(loclg, sbuf);

  // spectral resonator
  fft_fwd_k<<<dim3(65536), blk, 0, stream>>>(impsq, noise, freq);
  scan_res_k<<<dim3((NB * NCB + 255) / 256), blk, 0, stream>>>(freq, tco);
  fft_inv_k<<<dim3(65536), blk, 0, stream>>>(freq, frames);
  overlap_add_k<<<dim3((unsigned)(((size_t)NB * NSAMP + 255) / 256)), blk, 0, stream>>>(frames, sig);
  shift_out_k<<<dim3((unsigned)(((size_t)NB * NSAMP + 255) / 256)), blk, 0, stream>>>(sig, sbuf, (float*)d_out);
  (void)out_size; (void)n_in;
}
